// Attention_56186762166652
// MI455X (gfx1250) — compile-verified, matrix operand order checked by reading, not run on hardware
//
#include <hip/hip_runtime.h>

// ---------------------------------------------------------------------------
// Causal multi-head attention forward for MI455X (gfx1250, wave32, WMMA).
//   B=4, T=2048, D=1024, H=16, Dh=64
// Matrix math: v_wmma_f32_16x16x32_f16.
// Data movement: global_load_async_to_lds_b128 (ASYNCcnt) with double-buffered
// LDS tiles so DMA of tile i+1 overlaps WMMA on tile i.
// ---------------------------------------------------------------------------

typedef __attribute__((ext_vector_type(16))) _Float16 v16h;
typedef __attribute__((ext_vector_type(8)))  float    v8f;

#define Bsz    4
#define Tsz    2048
#define Dm     1024
#define Hn     16
#define Dh     64
#define ROWS   (Bsz * Tsz)        // 8192

static __device__ __forceinline__ v8f wmma_f16(v16h a, v16h b, v8f c) {
  return __builtin_amdgcn_wmma_f32_16x16x32_f16(
      /*neg_a=*/false, a, /*neg_b=*/false, b,
      /*c_mod=*/(short)0, c, /*reuse_a=*/false, /*reuse_b=*/false);
}

// Async DMA: copy 16 bytes (per lane) from global memory directly into LDS.
// VDST VGPR carries the LDS byte offset (= low 32 bits of the generic pointer,
// per ISA aperture rules). Tracked by ASYNCcnt, drained with s_wait_asynccnt.
static __device__ __forceinline__ void async_copy16(void* ldsDst, const void* src) {
  asm volatile("global_load_async_to_lds_b128 %0, %1, off"
               :: "v"((unsigned int)(size_t)ldsDst),
                  "v"((unsigned long long)(size_t)src)
               : "memory");
}
#define WAIT_ASYNC_0() asm volatile("s_wait_asynccnt 0x0" ::: "memory")
#define WAIT_ASYNC_4() asm volatile("s_wait_asynccnt 0x4" ::: "memory")
#define WAIT_ASYNC_8() asm volatile("s_wait_asynccnt 0x8" ::: "memory")

// Load a 16x32 f16 A/B fragment from an LDS tile (rowStride in halfs).
// ISA 7.12.2: lane (l&15) = matrix row; halfs 0..7  <-> K = kBase + hi*8 + 0..7
//             halfs 8..15 <-> K = kBase + 16 + hi*8 + 0..7
// -> exactly two contiguous 16-byte LDS reads (ds_load_b128) per lane.
static __device__ __forceinline__ v16h load_frag(const _Float16* tile, int rowBase,
                                                 int rowStride, int kBase) {
  const int lane = threadIdx.x & 31;
  const int r    = rowBase + (lane & 15);
  const int hi   = lane >> 4;
  const _Float16* p = tile + r * rowStride + kBase + hi * 8;
  v16h f;
  *((uint4*)&f)       = *(const uint4*)(p);
  *(((uint4*)&f) + 1) = *(const uint4*)(p + 16);
  return f;
}

// ---------------------------------------------------------------------------
// fp32 -> f16 conversion
// ---------------------------------------------------------------------------
__global__ void cvt_f32_to_f16(const float* __restrict__ src,
                               _Float16* __restrict__ dst, int n) {
  int i = blockIdx.x * blockDim.x + threadIdx.x;
  if (i < n) dst[i] = (_Float16)src[i];
}

// ---------------------------------------------------------------------------
// GEMM: Y[r,c] = sum_k A[r,k] * W[c,k] + bias[c]   (A: MxK, W: NxK, row-major)
// Block: 128 thr = 4 waves (2x2), tile 128(M) x 128(N), K-step 32.
// Wave tile 64x64 -> 16 WMMA per K-step, 16 accumulator v8f.
// Double-buffered LDS fed by async-to-LDS DMA (8 async insts / thread / tile).
// mode 0/1: f16 out, head layout [B,H,T,Dh]   (Q, K)
// mode 2  : f16 out, transposed  [B,H,Dh,T]   (V)
// mode 3  : f32 out, row-major   [B*T, D]     (final projection -> d_out)
// ---------------------------------------------------------------------------
__global__ __launch_bounds__(128) void gemm_wmma_f16(
    const _Float16* __restrict__ A, const _Float16* __restrict__ W,
    const float* __restrict__ bias, int M, int N, int K, int mode,
    void* __restrict__ outp) {
  __shared__ __align__(16) _Float16 sA[2][128 * 32];
  __shared__ __align__(16) _Float16 sB[2][128 * 32];

  const int tid   = threadIdx.x;
  const int wave  = tid >> 5;
  const int lane  = tid & 31;
  const int waveM = wave >> 1;            // 0..1
  const int waveN = wave & 1;             // 0..1
  const int blockM = blockIdx.x * 128;
  const int blockN = blockIdx.y * 128;

  // Issue one K-step tile (A: 512 uint4, B: 512 uint4) as async DMA.
  auto stage = [&](int buf, int k0) {
    #pragma unroll
    for (int i = tid; i < 512; i += 128) {          // 4 insts
      int r = i >> 2, ch = i & 3;
      async_copy16(&((uint4*)sA[buf])[i],
                   A + (size_t)(blockM + r) * K + k0 + ch * 8);
    }
    #pragma unroll
    for (int i = tid; i < 512; i += 128) {          // 4 insts
      int r = i >> 2, ch = i & 3;
      async_copy16(&((uint4*)sB[buf])[i],
                   W + (size_t)(blockN + r) * K + k0 + ch * 8);
    }
  };

  v8f c[4][4] = {};
  stage(0, 0);

  for (int k0 = 0; k0 < K; k0 += 32) {
    const int b = (k0 >> 5) & 1;
    if (k0 + 32 < K) {
      stage(b ^ 1, k0 + 32);   // overlap DMA of next tile with compute
      WAIT_ASYNC_8();          // drain only the older tile's 8 copies
    } else {
      WAIT_ASYNC_0();
    }
    __syncthreads();

    v16h af[4];
    #pragma unroll
    for (int mi = 0; mi < 4; ++mi)
      af[mi] = load_frag(sA[b], waveM * 64 + mi * 16, 32, 0);
    #pragma unroll
    for (int ni = 0; ni < 4; ++ni) {
      v16h bf = load_frag(sB[b], waveN * 64 + ni * 16, 32, 0);
      #pragma unroll
      for (int mi = 0; mi < 4; ++mi)
        c[mi][ni] = wmma_f16(af[mi], bf, c[mi][ni]);
    }
    __syncthreads();           // all reads of buf done before it is re-staged
  }

  // Epilogue. C layout (ISA 7.12.2): VGPR j -> row mOff+j, col = lane&15.
  const int n    = lane & 15;
  const int mOff = (lane >> 4) * 8;
  #pragma unroll
  for (int mi = 0; mi < 4; ++mi) {
    #pragma unroll
    for (int ni = 0; ni < 4; ++ni) {
      #pragma unroll
      for (int j = 0; j < 8; ++j) {
        int row = blockM + waveM * 64 + mi * 16 + mOff + j;
        int col = blockN + waveN * 64 + ni * 16 + n;
        float v = c[mi][ni][j] + bias[col];
        if (mode == 3) {
          ((float*)outp)[(size_t)row * N + col] = v;
        } else {
          int b2 = row >> 11, t = row & 2047;   // row = b*T + t
          int h  = col >> 6,  d = col & 63;     // col = h*Dh + d
          _Float16* o16 = (_Float16*)outp;
          size_t bh = (size_t)(b2 * Hn + h);
          if (mode == 2)
            o16[(bh * Dh + d) * Tsz + t] = (_Float16)v;   // V^T: [B,H,Dh,T]
          else
            o16[(bh * Tsz + t) * Dh + d] = (_Float16)v;   // Q/K: [B,H,T,Dh]
        }
      }
    }
  }
}

// ---------------------------------------------------------------------------
// Flash attention (causal). Block = 4 waves (128 thr), one (b,h) + 64 q-rows.
// Wave w owns q rows [qb + 16w, qb + 16w + 15]. Stream 32 keys / iteration,
// K/V tiles double-buffered via async-to-LDS DMA (4 async insts / thread).
// ---------------------------------------------------------------------------
__global__ __launch_bounds__(128) void flash_attn_wmma(
    const _Float16* __restrict__ Qh, const _Float16* __restrict__ Kh,
    const _Float16* __restrict__ Vt, _Float16* __restrict__ attnOut) {
  __shared__ __align__(16) _Float16 sQ[64 * 64];       // 64 q-rows x 64 d
  __shared__ __align__(16) _Float16 sK[2][32 * 64];    // 32 keys  x 64 d
  __shared__ __align__(16) _Float16 sV[2][64 * 32];    // 64 d     x 32 keys
  __shared__ __align__(16) _Float16 sP[4][16 * 32];

  const int tid  = threadIdx.x;
  const int wave = tid >> 5;
  const int lane = tid & 31;
  const int bh   = blockIdx.y;                 // b*16 + h
  const int qb   = blockIdx.x * 64;

  const _Float16* Qbase = Qh + (size_t)bh * Tsz * Dh;
  const _Float16* Kbase = Kh + (size_t)bh * Tsz * Dh;
  const _Float16* Vbase = Vt + (size_t)bh * Dh * Tsz;

  const int numKT = (qb >> 5) + 2;             // key tiles covering 0..qb+63

  auto stageKV = [&](int buf, int kb) {
    #pragma unroll
    for (int i = tid; i < 256; i += 128) {     // K tile: 2 insts
      int r = i >> 3, ch = i & 7;
      async_copy16(&((uint4*)sK[buf])[i],
                   Kbase + (size_t)(kb + r) * Dh + ch * 8);
    }
    #pragma unroll
    for (int i = tid; i < 256; i += 128) {     // V^T tile: 2 insts
      int r = i >> 2, ch = i & 3;
      async_copy16(&((uint4*)sV[buf])[i],
                   Vbase + (size_t)r * Tsz + kb + ch * 8);
    }
  };

  // Stage Q tile (4 insts) + first K/V tile (4 insts).
  #pragma unroll
  for (int i = tid; i < 512; i += 128) {
    int r = i >> 3, ch = i & 7;
    async_copy16(&((uint4*)sQ)[i], Qbase + (size_t)(qb + r) * Dh + ch * 8);
  }
  stageKV(0, 0);
  WAIT_ASYNC_4();                              // Q complete (K/V0 in flight)
  __syncthreads();
  const v16h qa0 = load_frag(sQ, wave * 16, 64, 0);    // d = 0..31
  const v16h qa1 = load_frag(sQ, wave * 16, 64, 32);   // d = 32..63

  v8f o[4] = {};
  float m[8], l[8];
  #pragma unroll
  for (int j = 0; j < 8; ++j) { m[j] = -3.0e38f; l[j] = 0.0f; }

  const int n        = lane & 15;
  const int mOff     = (lane >> 4) * 8;
  const int qRowBase = qb + wave * 16 + mOff;  // row of entry j = qRowBase + j

  for (int kt = 0; kt < numKT; ++kt) {
    const int kb = kt * 32;
    const int b  = kt & 1;
    if (kt + 1 < numKT) {
      stageKV(b ^ 1, kb + 32);                 // DMA next tile during compute
      WAIT_ASYNC_4();                          // drain current tile's copies
      if (tid < 32)                            // gfx1250 global_prefetch path
        __builtin_prefetch(Kbase + (size_t)(kb + 64 + tid) * Dh, 0, 1);
    } else {
      WAIT_ASYNC_0();
    }
    __syncthreads();

    // S = Q K^T (two 16x16 key tiles, K-dim = Dh = 64 -> 2 WMMA each)
    v8f s0 = {}, s1 = {};
    {
      v16h k0a = load_frag(sK[b], 0, 64, 0),  k0b = load_frag(sK[b], 0, 64, 32);
      v16h k1a = load_frag(sK[b], 16, 64, 0), k1b = load_frag(sK[b], 16, 64, 32);
      s0 = wmma_f16(qa0, k0a, s0);  s0 = wmma_f16(qa1, k0b, s0);
      s1 = wmma_f16(qa0, k1a, s1);  s1 = wmma_f16(qa1, k1b, s1);
    }

    // Scale + causal mask + online softmax (row reductions over 16 lanes).
    #pragma unroll
    for (int j = 0; j < 8; ++j) {
      const int row = qRowBase + j;
      float a = s0[j] * 0.125f;                // 1/sqrt(64)
      float bb = s1[j] * 0.125f;
      if (kb + n      > row) a  = -1.0e6f;
      if (kb + 16 + n > row) bb = -1.0e6f;
      float mx = fmaxf(a, bb);
      #pragma unroll
      for (int d = 1; d < 16; d <<= 1) mx = fmaxf(mx, __shfl_xor(mx, d, 32));
      const float mNew = fmaxf(m[j], mx);
      const float p0   = __expf(a - mNew);
      const float p1   = __expf(bb - mNew);
      float rs = p0 + p1;
      #pragma unroll
      for (int d = 1; d < 16; d <<= 1) rs += __shfl_xor(rs, d, 32);
      const float alpha = __expf(m[j] - mNew);
      l[j] = l[j] * alpha + rs;
      m[j] = mNew;
      #pragma unroll
      for (int nb = 0; nb < 4; ++nb) o[nb][j] *= alpha;   // rescale O row
      sP[wave][(mOff + j) * 32 + n]      = (_Float16)p0;  // C-layout -> A-layout
      sP[wave][(mOff + j) * 32 + 16 + n] = (_Float16)p1;
    }

    // O += P(16x32) x V(32x64): 4 WMMA, K = 32 keys. (LDS in-order per wave.)
    const v16h pf = load_frag(sP[wave], 0, 32, 0);
    #pragma unroll
    for (int nb = 0; nb < 4; ++nb) {
      v16h vf = load_frag(sV[b], nb * 16, 32, 0);
      o[nb] = wmma_f16(pf, vf, o[nb]);
    }
    __syncthreads();           // all reads of buf b done before it is re-staged
  }

  // Normalize and scatter to attnOut [B, T, D] (f16) for the output projection.
  const int b2 = bh >> 4, h = bh & 15;
  #pragma unroll
  for (int nb = 0; nb < 4; ++nb) {
    #pragma unroll
    for (int j = 0; j < 8; ++j) {
      const int t   = qRowBase + j;
      const int col = nb * 16 + n;
      const float v = o[nb][j] / l[j];
      attnOut[((size_t)b2 * Tsz + t) * Dm + h * Dh + col] = (_Float16)v;
    }
  }
}

// ---------------------------------------------------------------------------
// Host launch
// ---------------------------------------------------------------------------
extern "C" void kernel_launch(void* const* d_in, const int* in_sizes, int n_in,
                              void* d_out, int out_size, void* d_ws, size_t ws_size,
                              hipStream_t stream) {
  const float* x  = (const float*)d_in[0];
  const float* Wq = (const float*)d_in[1];
  const float* bq = (const float*)d_in[2];
  const float* Wk = (const float*)d_in[3];
  const float* bk = (const float*)d_in[4];
  const float* Wv = (const float*)d_in[5];
  const float* bv = (const float*)d_in[6];
  const float* Wo = (const float*)d_in[7];
  const float* bo = (const float*)d_in[8];

  char* ws = (char*)d_ws;
  const size_t MB = 1024 * 1024;
  _Float16* xh  = (_Float16*)(ws + 0);          // 16 MB: [8192,1024] f16
  _Float16* Wqh = (_Float16*)(ws + 16 * MB);    // 2 MB each
  _Float16* Wkh = (_Float16*)(ws + 18 * MB);
  _Float16* Wvh = (_Float16*)(ws + 20 * MB);
  _Float16* Woh = (_Float16*)(ws + 22 * MB);
  _Float16* Qh  = (_Float16*)(ws + 24 * MB);    // 16 MB: [B,H,T,Dh]
  _Float16* Kh  = (_Float16*)(ws + 40 * MB);    // 16 MB: [B,H,T,Dh]
  _Float16* Vt  = (_Float16*)(ws + 56 * MB);    // 16 MB: [B,H,Dh,T]
  _Float16* Ah  = (_Float16*)(ws + 72 * MB);    // 16 MB: [B,T,D]

  const int nx = ROWS * Dm;                     // 8388608
  const int nw = Dm * Dm;                       // 1048576
  cvt_f32_to_f16<<<(nx + 255) / 256, 256, 0, stream>>>(x,  xh,  nx);
  cvt_f32_to_f16<<<(nw + 255) / 256, 256, 0, stream>>>(Wq, Wqh, nw);
  cvt_f32_to_f16<<<(nw + 255) / 256, 256, 0, stream>>>(Wk, Wkh, nw);
  cvt_f32_to_f16<<<(nw + 255) / 256, 256, 0, stream>>>(Wv, Wvh, nw);
  cvt_f32_to_f16<<<(nw + 255) / 256, 256, 0, stream>>>(Wo, Woh, nw);

  dim3 gg(ROWS / 128, Dm / 128);                // 64 x 8
  gemm_wmma_f16<<<gg, 128, 0, stream>>>(xh, Wqh, bq, ROWS, Dm, Dm, 0, Qh);
  gemm_wmma_f16<<<gg, 128, 0, stream>>>(xh, Wkh, bk, ROWS, Dm, Dm, 1, Kh);
  gemm_wmma_f16<<<gg, 128, 0, stream>>>(xh, Wvh, bv, ROWS, Dm, Dm, 2, Vt);

  flash_attn_wmma<<<dim3(Tsz / 64, Bsz * Hn), 128, 0, stream>>>(Qh, Kh, Vt, Ah);

  gemm_wmma_f16<<<gg, 128, 0, stream>>>(Ah, Woh, bo, ROWS, Dm, Dm, 3, d_out);
}
